// CustomAttention_36352603193460
// MI455X (gfx1250) — compile-verified, matrix operand order checked by reading, not run on hardware
//
#include <hip/hip_runtime.h>
#include <math.h>

// Problem constants (match reference)
#define NN    4096      // nodes
#define INF_  512       // input feature size
#define DH    128       // per-head output dim
#define NH    4         // heads
#define HD    512       // NH*DH
#define SLOPE 0.2f
#define NEGV  (-1e30f)

typedef __attribute__((ext_vector_type(16))) _Float16 v16h;
typedef __attribute__((ext_vector_type(8)))  _Float16 v8h;
typedef __attribute__((ext_vector_type(8)))  float    v8f;

union FragH { v16h v; v8h q[2]; _Float16 h[16]; };
union FragF { v8f  v; float f[8]; };
union PackH8 { v8h v; _Float16 h[8]; };
union PackH2 { unsigned u; _Float16 h[2]; };

// ---------------------------------------------------------------------------
// f32 -> f16 convert (plain)
// ---------------------------------------------------------------------------
__global__ __launch_bounds__(256)
void cvt_f16_kernel(const float* __restrict__ src, _Float16* __restrict__ dst, int n) {
    int i = blockIdx.x * blockDim.x + threadIdx.x;
    if (i < n) dst[i] = (_Float16)src[i];
}

// f32 [rows x cols] -> f16 transposed [cols x rows]
__global__ __launch_bounds__(256)
void cvt_f16_t_kernel(const float* __restrict__ src, _Float16* __restrict__ dst,
                      int rows, int cols) {
    int idx = blockIdx.x * blockDim.x + threadIdx.x;
    if (idx < rows * cols) {
        int r = idx / cols, c = idx % cols;
        dst[(size_t)c * rows + r] = (_Float16)src[idx];
    }
}

// ---------------------------------------------------------------------------
// C[MxHD] = A16[MxK] @ Bt16[HDxK] (B pre-transposed, K-contiguous).
// Block = 8 waves; block tile 16 rows x 128 cols; wave tile 16x16.
// Whole 16x512 A slab staged in LDS ONCE (16KB of the 320KB WGP LDS), then
// the 16 WMMA K-steps run barrier-free: 2 ds_load_b128 + 2 global_load_b128
// + v_wmma each, fully unrolled for clausing/pipelining.
// Optionally writes a transposed f16 copy of C (column-major, one b128/lane).
// ---------------------------------------------------------------------------
__global__ __launch_bounds__(256)
void gemm_wmma_kernel(const _Float16* __restrict__ A, const _Float16* __restrict__ Bt,
                      float* __restrict__ C32, _Float16* __restrict__ C16T) {
    __shared__ _Float16 As[16][520];   // 1040B rows: 16B-aligned, conflict-free
    const int lane    = threadIdx.x & 31;
    const int wave    = threadIdx.x >> 5;
    const int half    = lane >> 4;
    const int mn      = lane & 15;
    const int rowBase = blockIdx.x * 16;
    const int colBase = blockIdx.y * 128 + wave * 16;

    // Stage entire A row-slab: 16 rows x 512 halves = 1024 x 16B chunks.
    for (int c = threadIdx.x; c < 1024; c += 256) {
        int r = c >> 6, q = (c & 63) * 8;
        *(v8h*)&As[r][q] = *(const v8h*)&A[(size_t)(rowBase + r) * INF_ + q];
    }
    __syncthreads();

    const _Float16* bcol = Bt + (size_t)(colBase + mn) * INF_;

    FragF acc;
#pragma unroll
    for (int r = 0; r < 8; ++r) acc.f[r] = 0.0f;

#pragma unroll
    for (int kk = 0; kk < INF_; kk += 32) {
        FragH fa, fb;
        fa.q[0] = *(const v8h*)&As[mn][kk + 8 * half];
        fa.q[1] = *(const v8h*)&As[mn][kk + 16 + 8 * half];
        fb.q[0] = *(const v8h*)&bcol[kk + 8 * half];
        fb.q[1] = *(const v8h*)&bcol[kk + 16 + 8 * half];
        acc.v = __builtin_amdgcn_wmma_f32_16x16x32_f16(
            false, fa.v, false, fb.v, (short)0, acc.v, false, false);
    }

#pragma unroll
    for (int r = 0; r < 8; ++r)
        C32[(size_t)(rowBase + r + 8 * half) * HD + colBase + mn] = acc.f[r];

    if (C16T) {   // column-major: 8 consecutive rows per lane -> one b128 store
        PackH8 pk;
#pragma unroll
        for (int r = 0; r < 8; ++r) pk.h[r] = (_Float16)acc.f[r];
        *(v8h*)&C16T[(size_t)(colBase + mn) * NN + rowBase + 8 * half] = pk.v;
    }
}

// ---------------------------------------------------------------------------
// f1[h,n] = support[h,n,:]·wu[h];  f2[h,n] = support[h,n,:]·wv[h]
// One wave32 per (h,n); D=128 = 32 lanes x float4; shuffle reduction.
// ---------------------------------------------------------------------------
__global__ __launch_bounds__(256)
void fvec_kernel(const float* __restrict__ support, const float* __restrict__ wu,
                 const float* __restrict__ wv, float* __restrict__ f1,
                 float* __restrict__ f2) {
    int wid  = blockIdx.x * 8 + (threadIdx.x >> 5);  // 0..NH*NN-1
    int lane = threadIdx.x & 31;
    int h = wid & (NH - 1);
    int n = wid >> 2;
    const float4 sv = *(const float4*)(support + (size_t)n * HD + h * DH + lane * 4);
    const float4 uv = *(const float4*)(wu + h * DH + lane * 4);
    const float4 vv = *(const float4*)(wv + h * DH + lane * 4);
    float s1 = sv.x * uv.x + sv.y * uv.y + sv.z * uv.z + sv.w * uv.w;
    float s2 = sv.x * vv.x + sv.y * vv.y + sv.z * vv.z + sv.w * vv.w;
    for (int off = 16; off > 0; off >>= 1) {
        s1 += __shfl_xor(s1, off, 32);
        s2 += __shfl_xor(s2, off, 32);
    }
    if (lane == 0) { f1[h * NN + n] = s1; f2[h * NN + n] = s2; }
}

// ---------------------------------------------------------------------------
// Online softmax stats per (h,i) row over finite-masked logits.
// One wave32 per row, float4 streaming; adjacency stays hot in 192MB L2.
// ---------------------------------------------------------------------------
__global__ __launch_bounds__(256)
void rowstats_kernel(const float* __restrict__ adj, const float* __restrict__ f1,
                     const float* __restrict__ f2, float* __restrict__ rowmax,
                     float* __restrict__ rowsum) {
    int wid  = blockIdx.x * 8 + (threadIdx.x >> 5);
    int lane = threadIdx.x & 31;
    int h = wid & (NH - 1);
    int i = wid >> 2;
    const float f2i = f2[h * NN + i];
    const float4* arow = (const float4*)(adj + (size_t)i * NN);
    const float4* f1h  = (const float4*)(f1 + h * NN);

    float m = -INFINITY, s = 0.f;
    for (int jb = lane; jb < NN / 4; jb += 32) {
        float4 a4 = arow[jb];
        float4 g4 = f1h[jb];
        const float av[4] = {a4.x, a4.y, a4.z, a4.w};
        const float gv[4] = {g4.x, g4.y, g4.z, g4.w};
#pragma unroll
        for (int u = 0; u < 4; ++u) {
            float w  = gv[u] + f2i;
            float l  = (w >= 0.f) ? w : SLOPE * w;
            float mk = l * av[u];
            float z  = (mk != 0.f) ? mk : NEGV;
            if (z > m) { s = s * __expf(m - z) + 1.f; m = z; }
            else       { s += __expf(z - m); }
        }
    }
    for (int off = 16; off > 0; off >>= 1) {
        float mo = __shfl_xor(m, off, 32);
        float so = __shfl_xor(s, off, 32);
        float mx = fmaxf(m, mo);
        s = s * __expf(m - mx) + so * __expf(mo - mx);
        m = mx;
    }
    if (lane == 0) { rowmax[h * NN + i] = m; rowsum[h * NN + i] = s; }
}

// ---------------------------------------------------------------------------
// out[i, h*DH+d] = sum_j attn[h,i,j]*support[h,j,d] + bias + proj + proj_b
// Block = 8 waves, 16 rows x full head-D. 16x32 probability tile computed in
// f32 (float2 reads, packed b32 LDS stores) into a DOUBLE-BUFFERED LDS tile
// (one barrier per j-step), consumed as the shared WMMA A fragment; B frags
// from transposed support f16 (j-contiguous -> b128 loads).
// ---------------------------------------------------------------------------
__global__ __launch_bounds__(256)
void attn_av_kernel(const float* __restrict__ adj, const float* __restrict__ f1,
                    const float* __restrict__ f2, const float* __restrict__ rowmax,
                    const float* __restrict__ rowsum, const _Float16* __restrict__ supT,
                    const float* __restrict__ proj, const float* __restrict__ bias,
                    const float* __restrict__ proj_b, float* __restrict__ out) {
    __shared__ _Float16 Ps[2][16][40];
    __shared__ float f2s[16], ms[16], invs[16];

    const int h       = blockIdx.y;
    const int rowBase = blockIdx.x * 16;
    const int lane    = threadIdx.x & 31;
    const int wave    = threadIdx.x >> 5;
    const int half    = lane >> 4;
    const int mn      = lane & 15;
    const int dBase   = wave * 16;

    if (threadIdx.x < 16) {
        int i = rowBase + threadIdx.x;
        f2s[threadIdx.x] = f2[h * NN + i];
        ms[threadIdx.x]  = rowmax[h * NN + i];
        float s = rowsum[h * NN + i];
        invs[threadIdx.x] = (s > 0.f) ? (1.f / s) : 0.f;  // edgeless rows -> zeros
    }
    __syncthreads();

    FragF acc;
#pragma unroll
    for (int r = 0; r < 8; ++r) acc.f[r] = 0.0f;

    const float* f1h = f1 + h * NN;
    const _Float16* bcol = supT + (size_t)(h * DH + dBase + mn) * NN;

    const int pi = threadIdx.x >> 4;        // probability tile row 0..15
    const int pj = (threadIdx.x & 15) * 2;  // probability tile col pair

    for (int jb = 0; jb < NN; jb += 32) {
        const int buf = (jb >> 5) & 1;
        {
            int i = rowBase + pi, j = jb + pj;
            if (jb + 32 < NN)
                __builtin_prefetch(&adj[(size_t)i * NN + j + 32], 0, 0);
            float2 a2 = *(const float2*)&adj[(size_t)i * NN + j];
            float2 g2 = *(const float2*)&f1h[j];
            float w0  = g2.x + f2s[pi];
            float w1  = g2.y + f2s[pi];
            float l0  = (w0 >= 0.f) ? w0 : SLOPE * w0;
            float l1  = (w1 >= 0.f) ? w1 : SLOPE * w1;
            float mk0 = l0 * a2.x;
            float mk1 = l1 * a2.y;
            PackH2 pk;
            pk.h[0] = (_Float16)((mk0 != 0.f) ? __expf(mk0 - ms[pi]) * invs[pi] : 0.f);
            pk.h[1] = (_Float16)((mk1 != 0.f) ? __expf(mk1 - ms[pi]) * invs[pi] : 0.f);
            *(unsigned*)&Ps[buf][pi][pj] = pk.u;
        }
        __syncthreads();   // producers done; next iteration writes buf^1, so
                           // no second barrier is needed before reuse.

        FragH fa, fb;
        fa.q[0] = *(const v8h*)&Ps[buf][mn][8 * half];
        fa.q[1] = *(const v8h*)&Ps[buf][mn][16 + 8 * half];
        fb.q[0] = *(const v8h*)&bcol[jb + 8 * half];
        fb.q[1] = *(const v8h*)&bcol[jb + 16 + 8 * half];

        acc.v = __builtin_amdgcn_wmma_f32_16x16x32_f16(
            false, fa.v, false, fb.v, (short)0, acc.v, false, false);
    }

#pragma unroll
    for (int r = 0; r < 8; ++r) {
        int i   = rowBase + r + 8 * half;
        int col = h * DH + dBase + mn;
        out[(size_t)i * HD + col] =
            acc.f[r] + bias[col] + proj[(size_t)i * HD + col] + proj_b[col];
    }
}

// ---------------------------------------------------------------------------
// Host launcher
// ---------------------------------------------------------------------------
extern "C" void kernel_launch(void* const* d_in, const int* in_sizes, int n_in,
                              void* d_out, int out_size, void* d_ws, size_t ws_size,
                              hipStream_t stream) {
    const float* inputs = (const float*)d_in[0];   // [N, IN]
    const float* adj    = (const float*)d_in[1];   // [N, N]
    const float* weight = (const float*)d_in[2];   // [IN, HD]
    const float* wu     = (const float*)d_in[3];   // [H, D, 1]
    const float* wv     = (const float*)d_in[4];   // [H, D, 1]
    const float* bias   = (const float*)d_in[5];   // [1, HD]
    const float* proj_w = (const float*)d_in[6];   // [IN, HD]
    const float* proj_b = (const float*)d_in[7];   // [HD]
    float* out = (float*)d_out;                    // [N, HD]

    // Workspace carve-up (~25.5 MB total)
    char* w = (char*)d_ws;
    auto take = [&](size_t bytes) { char* p = w; w += (bytes + 255) & ~(size_t)255; return p; };
    _Float16* Xh   = (_Float16*)take((size_t)NN * INF_ * sizeof(_Float16));
    _Float16* Wt   = (_Float16*)take((size_t)INF_ * HD * sizeof(_Float16));  // [HD][IN]
    _Float16* Pt   = (_Float16*)take((size_t)INF_ * HD * sizeof(_Float16));  // [HD][IN]
    _Float16* SupT = (_Float16*)take((size_t)NN * HD * sizeof(_Float16));    // [HD][NN]
    float*    S32  = (float*)take((size_t)NN * HD * sizeof(float));
    float*    P32  = (float*)take((size_t)NN * HD * sizeof(float));
    float*    f1   = (float*)take((size_t)NH * NN * sizeof(float));
    float*    f2   = (float*)take((size_t)NH * NN * sizeof(float));
    float*    rm   = (float*)take((size_t)NH * NN * sizeof(float));
    float*    rs   = (float*)take((size_t)NH * NN * sizeof(float));

    // 1) f16 operands: X plain; W/P transposed (K-contiguous for WMMA B frags)
    cvt_f16_kernel<<<(NN * INF_ + 255) / 256, 256, 0, stream>>>(inputs, Xh, NN * INF_);
    cvt_f16_t_kernel<<<(INF_ * HD + 255) / 256, 256, 0, stream>>>(weight, Wt, INF_, HD);
    cvt_f16_t_kernel<<<(INF_ * HD + 255) / 256, 256, 0, stream>>>(proj_w, Pt, INF_, HD);

    // 2) support = X@W (f32 + transposed f16); proj = X@P (f32)
    dim3 ggrid(NN / 16, HD / 128);
    gemm_wmma_kernel<<<ggrid, 256, 0, stream>>>(Xh, Wt, S32, SupT);
    gemm_wmma_kernel<<<ggrid, 256, 0, stream>>>(Xh, Pt, P32, nullptr);

    // 3) per-node attention score halves f1, f2
    fvec_kernel<<<(NH * NN) / 8, 256, 0, stream>>>(S32, wu, wv, f1, f2);

    // 4) row softmax stats
    rowstats_kernel<<<(NH * NN) / 8, 256, 0, stream>>>(adj, f1, f2, rm, rs);

    // 5) attn @ support fused with bias/residual epilogue
    attn_av_kernel<<<dim3(NN / 16, NH), 256, 0, stream>>>(
        adj, f1, f2, rm, rs, SupT, P32, bias, proj_b, out);
}
